// GCN_86045374808468
// MI455X (gfx1250) — matrix operand order repaired and checked by reading, not compiled, not run on hardware
//
#include <hip/hip_runtime.h>
#include <math.h>

// ---------------------------------------------------------------------------
// CDNA5 (gfx1250) GCN forward: bf16 WMMA GEMMs + L2 fp32-atomic aggregation.
// ---------------------------------------------------------------------------

typedef __attribute__((ext_vector_type(16))) __bf16 v16bf;
typedef __attribute__((ext_vector_type(8)))  float  v8f;

union Frag {
    v16bf v;
    uint4 q[2];
};

__device__ __forceinline__ unsigned short f32_to_bf16(float f) {
    unsigned int u = __float_as_uint(f);
    u += 0x7FFFu + ((u >> 16) & 1u);   // round-to-nearest-even
    return (unsigned short)(u >> 16);
}

// ---------------------------------------------------------------------------
// Utility kernels
// ---------------------------------------------------------------------------
__global__ void k_zero(float* __restrict__ p, long n) {
    long gid = (long)blockIdx.x * blockDim.x + threadIdx.x;
    if (gid < n) p[gid] = 0.0f;
}

__global__ void k_degree(const int* __restrict__ dst, float* __restrict__ deg, int E) {
    long gid = (long)blockIdx.x * blockDim.x + threadIdx.x;
    if (gid < E) atomicAdd(&deg[dst[gid]], 1.0f);
}

// deg -> dis = rsqrt(deg + 1) (the +1 is the self-loop; always > 0)
__global__ void k_to_dis(float* __restrict__ deg, int N) {
    long gid = (long)blockIdx.x * blockDim.x + threadIdx.x;
    if (gid < N) deg[gid] = rsqrtf(deg[gid] + 1.0f);
}

// ---------------------------------------------------------------------------
// Pack W (K x Nout, f32 row-major) into WMMA B-fragment order, bf16.
// Layout: Wp[((t*NT + nt)*32 + lane)*16 + i], with
//   K = t*32 + 8*(lane>=16) + (i&7) + 16*(i>=8),  N = nt*16 + (lane&15)
// so each lane's 16-element fragment is one contiguous 32-byte run.
// ---------------------------------------------------------------------------
__global__ void k_pack_w(const float* __restrict__ W, int K, int Nout, int NT, int Kt,
                         unsigned short* __restrict__ Wp) {
    long gid = (long)blockIdx.x * blockDim.x + threadIdx.x;
    long total = (long)Kt * NT * 512;
    if (gid >= total) return;
    int i    = (int)(gid & 15);
    int lane = (int)((gid >> 4) & 31);
    int nt   = (int)((gid >> 9) % NT);
    int t    = (int)(gid / (512L * NT));
    int k = t * 32 + ((lane >> 4) << 3) + (i & 7) + ((i >> 3) << 4);
    int n = nt * 16 + (lane & 15);
    float v = (k < K) ? W[(size_t)k * Nout + n] : 0.0f;
    Wp[gid] = f32_to_bf16(v);
}

// ---------------------------------------------------------------------------
// WMMA GEMM: 32 x (NT*16) output tile per block, software-pipelined.
//   A: (M x K) f32 row-major, staged in LDS as bf16 (K padded to Kt*32).
//   Wp: pre-packed bf16 fragments (see k_pack_w).
//   blockDim.x = NT*32, wave w handles n-tile w with TWO accumulators
//   (row halves 0-15 / 16-31) sharing one B fragment.
// ---------------------------------------------------------------------------
#define LDS_STRIDE 520   // ushorts; breaks LDS bank alignment between rows

template <int Kt, int NT>
__global__ void __launch_bounds__(NT * 32)
k_wmma_gemm(const float* __restrict__ A, int M, int K,
            const unsigned short* __restrict__ Wp,
            float* __restrict__ Out, int ldOut) {
    __shared__ __align__(16) unsigned short lA[32 * LDS_STRIDE];

    const int tid      = threadIdx.x;
    const int nthreads = NT * 32;
    const int rowbase  = blockIdx.x * 32;
    const int Kpad     = Kt * 32;

    // Speculative prefetch of the next block's A tile toward L2
    // (emits global_prefetch_b8; dropped harmlessly if OOB).
    if (blockIdx.x + 1 < gridDim.x) {
        const char* nxt = (const char*)(A + (size_t)(rowbase + 32) * K);
        const size_t tilebytes = (size_t)32 * K * sizeof(float);
        for (size_t p = (size_t)tid * 128; p < tilebytes; p += (size_t)nthreads * 128)
            __builtin_prefetch(nxt + p, 0, 1);
    }

    // Cooperatively stage 32 x Kpad tile of A as bf16 (zero-pad K tail / OOB rows).
    for (int idx = tid; idx < 32 * Kpad; idx += nthreads) {
        int m = idx / Kpad;
        int k = idx - m * Kpad;
        int row = rowbase + m;
        float v = (k < K && row < M) ? A[(size_t)row * K + k] : 0.0f;
        lA[m * LDS_STRIDE + k] = f32_to_bf16(v);
    }
    __syncthreads();

    const int wave = tid >> 5;     // n-tile index
    const int lane = tid & 31;
    const int hi   = lane >> 4;    // 0: lanes 0-15, 1: lanes 16-31
    const int m    = lane & 15;    // A-fragment row within half-tile

    const unsigned short* wp = Wp + ((size_t)wave * 32 + lane) * 16;
    const unsigned short* la0 = &lA[m * LDS_STRIDE + hi * 8];
    const unsigned short* la1 = &lA[(m + 16) * LDS_STRIDE + hi * 8];

    auto load_frags = [&](int t, Frag& fb, Frag& fa0, Frag& fa1) {
        const int kbase = t * 32;
        const unsigned short* wpt = wp + (size_t)t * NT * 512;
        fb.q[0]  = *(const uint4*)(wpt);
        fb.q[1]  = *(const uint4*)(wpt + 8);
        fa0.q[0] = *(const uint4*)(la0 + kbase);
        fa0.q[1] = *(const uint4*)(la0 + kbase + 16);
        fa1.q[0] = *(const uint4*)(la1 + kbase);
        fa1.q[1] = *(const uint4*)(la1 + kbase + 16);
    };

    v8f acc0 = {}, acc1 = {};
    Frag fb, fa0, fa1;
    load_frags(0, fb, fa0, fa1);

#pragma unroll
    for (int t = 0; t < Kt - 1; ++t) {
        Frag nb, na0, na1;
        load_frags(t + 1, nb, na0, na1);   // in flight while WMMAs of step t issue
        acc0 = __builtin_amdgcn_wmma_f32_16x16x32_bf16(false, fa0.v, false, fb.v,
                                                       (short)0, acc0, false, false);
        acc1 = __builtin_amdgcn_wmma_f32_16x16x32_bf16(false, fa1.v, false, fb.v,
                                                       (short)0, acc1, false, false);
        fb = nb; fa0 = na0; fa1 = na1;
    }
    acc0 = __builtin_amdgcn_wmma_f32_16x16x32_bf16(false, fa0.v, false, fb.v,
                                                   (short)0, acc0, false, false);
    acc1 = __builtin_amdgcn_wmma_f32_16x16x32_bf16(false, fa1.v, false, fb.v,
                                                   (short)0, acc1, false, false);

    // C/D layout: VGPR r -> M = r + 8*hi, N = lane&15
    const int col = wave * 16 + (lane & 15);
#pragma unroll
    for (int r = 0; r < 8; ++r) {
        int row0 = rowbase + hi * 8 + r;
        int row1 = row0 + 16;
        if (row0 < M) Out[(size_t)row0 * ldOut + col] = acc0[r];
        if (row1 < M) Out[(size_t)row1 * ldOut + col] = acc1[r];
    }
}

// ---------------------------------------------------------------------------
// Edge aggregation: Agg[dst] += H[src] * dis[src]*dis[dst]
// one thread per (edge, 4-column chunk); F multiple of 4.
// ---------------------------------------------------------------------------
__global__ void k_edge_agg(const int* __restrict__ src, const int* __restrict__ dst,
                           const float* __restrict__ dis, const float* __restrict__ H,
                           float* __restrict__ Agg, int E, int F) {
    const int F4 = F >> 2;
    long gid = (long)blockIdx.x * blockDim.x + threadIdx.x;
    if (gid >= (long)E * F4) return;
    int e = (int)(gid / F4);
    int c = (int)(gid % F4) << 2;
    int s = src[e], d = dst[e];
    float w = dis[s] * dis[d];
    float4 v = *(const float4*)&H[(size_t)s * F + c];
    float* a = &Agg[(size_t)d * F + c];
    atomicAdd(a + 0, v.x * w);
    atomicAdd(a + 1, v.y * w);
    atomicAdd(a + 2, v.z * w);
    atomicAdd(a + 3, v.w * w);
}

// F = 3 variant, one thread per edge
__global__ void k_edge_agg3(const int* __restrict__ src, const int* __restrict__ dst,
                            const float* __restrict__ dis, const float* __restrict__ H,
                            float* __restrict__ Agg, int E) {
    long gid = (long)blockIdx.x * blockDim.x + threadIdx.x;
    if (gid >= E) return;
    int s = src[gid], d = dst[gid];
    float w = dis[s] * dis[d];
    const float* h = &H[(size_t)s * 3];
    float* a = &Agg[(size_t)d * 3];
    atomicAdd(a + 0, h[0] * w);
    atomicAdd(a + 1, h[1] * w);
    atomicAdd(a + 2, h[2] * w);
}

// ---------------------------------------------------------------------------
// Epilogue: H = relu(Agg + H*dis^2 (self-loop) + b)  (in place into H)
// ---------------------------------------------------------------------------
__global__ void k_bias_relu(const float* __restrict__ Agg, float* __restrict__ H,
                            const float* __restrict__ dis, const float* __restrict__ b,
                            long N, int F) {
    long gid = (long)blockIdx.x * blockDim.x + threadIdx.x;
    if (gid >= N * F) return;
    long n = gid / F;
    int  c = (int)(gid - n * F);
    float s = dis[n] * dis[n];
    float v = Agg[gid] + H[gid] * s + b[c];
    H[gid] = fmaxf(v, 0.0f);
}

// ---------------------------------------------------------------------------
// Layer 3 dense (32 -> 3), one node per thread
// ---------------------------------------------------------------------------
__global__ void k_dense3(const float* __restrict__ H, const float* __restrict__ W3,
                         float* __restrict__ H3, int N) {
    long n = (long)blockIdx.x * blockDim.x + threadIdx.x;
    if (n >= N) return;
    float a0 = 0.f, a1 = 0.f, a2 = 0.f;
    const float* h = &H[(size_t)n * 32];
#pragma unroll
    for (int k = 0; k < 32; ++k) {
        float hv = h[k];
        a0 += hv * W3[k * 3 + 0];
        a1 += hv * W3[k * 3 + 1];
        a2 += hv * W3[k * 3 + 2];
    }
    H3[n * 3 + 0] = a0;
    H3[n * 3 + 1] = a1;
    H3[n * 3 + 2] = a2;
}

// Final: out = log_softmax(Agg3 + H3*dis^2 + b3)
__global__ void k_finalize(const float* __restrict__ Agg3, const float* __restrict__ H3,
                           const float* __restrict__ dis, const float* __restrict__ b3,
                           float* __restrict__ out, int N) {
    long n = (long)blockIdx.x * blockDim.x + threadIdx.x;
    if (n >= N) return;
    float s = dis[n] * dis[n];
    float v0 = Agg3[n * 3 + 0] + H3[n * 3 + 0] * s + b3[0];
    float v1 = Agg3[n * 3 + 1] + H3[n * 3 + 1] * s + b3[1];
    float v2 = Agg3[n * 3 + 2] + H3[n * 3 + 2] * s + b3[2];
    float m = fmaxf(v0, fmaxf(v1, v2));
    float lse = logf(expf(v0 - m) + expf(v1 - m) + expf(v2 - m)) + m;
    out[n * 3 + 0] = v0 - lse;
    out[n * 3 + 1] = v1 - lse;
    out[n * 3 + 2] = v2 - lse;
}

// ---------------------------------------------------------------------------
// Launch
// ---------------------------------------------------------------------------
static inline int nblk(long total, int bs) { return (int)((total + bs - 1) / bs); }

extern "C" void kernel_launch(void* const* d_in, const int* in_sizes, int n_in,
                              void* d_out, int out_size, void* d_ws, size_t ws_size,
                              hipStream_t stream) {
    const int F_IN = 500, F1 = 128, F2 = 32, F3 = 3;
    const float* x  = (const float*)d_in[0];
    const int*   ei = (const int*)d_in[1];
    const float* W1 = (const float*)d_in[2];
    const float* b1 = (const float*)d_in[3];
    const float* W2 = (const float*)d_in[4];
    const float* b2 = (const float*)d_in[5];
    const float* W3 = (const float*)d_in[6];
    const float* b3 = (const float*)d_in[7];
    float* out = (float*)d_out;

    const int N = in_sizes[0] / F_IN;    // 100000
    const int E = in_sizes[1] / 2;       // 1600000
    const int* src = ei;                 // edge_index[0]
    const int* dst = ei + E;             // edge_index[1]

    // ---- workspace carving (256-byte aligned) ----
    char* ws = (char*)d_ws;
    size_t off = 0;
    auto carve = [&](size_t bytes) -> char* {
        char* p = ws + off;
        off += (bytes + 255) & ~(size_t)255;
        return p;
    };
    float* dis  = (float*)carve((size_t)N * 4);                 // deg, then rsqrt in place
    float* h1   = (float*)carve((size_t)N * F1 * 4);
    float* agg1 = (float*)carve((size_t)N * F1 * 4);
    float* h2   = (float*)carve((size_t)N * F2 * 4);
    float* agg2 = (float*)carve((size_t)N * F2 * 4);
    float* h3   = (float*)carve((size_t)N * F3 * 4);
    float* agg3 = (float*)carve((size_t)N * F3 * 4);
    const int Kt1 = (F_IN + 31) / 32, NT1 = F1 / 16;            // 16, 8
    const int Kt2 = F1 / 32,          NT2 = F2 / 16;            // 4, 2
    unsigned short* Wp1 = (unsigned short*)carve((size_t)Kt1 * NT1 * 512 * 2);
    unsigned short* Wp2 = (unsigned short*)carve((size_t)Kt2 * NT2 * 512 * 2);
    (void)ws_size;

    const int BS = 256;

    // ---- zero accumulators / degree ----
    k_zero<<<nblk(N, BS), BS, 0, stream>>>(dis, N);
    k_zero<<<nblk((long)N * F1, BS), BS, 0, stream>>>(agg1, (long)N * F1);
    k_zero<<<nblk((long)N * F2, BS), BS, 0, stream>>>(agg2, (long)N * F2);
    k_zero<<<nblk((long)N * F3, BS), BS, 0, stream>>>(agg3, (long)N * F3);

    // ---- degrees -> dis ----
    k_degree<<<nblk(E, BS), BS, 0, stream>>>(dst, dis, E);
    k_to_dis<<<nblk(N, BS), BS, 0, stream>>>(dis, N);

    // ---- pack weights into WMMA fragment order (bf16) ----
    k_pack_w<<<nblk((long)Kt1 * NT1 * 512, BS), BS, 0, stream>>>(W1, F_IN, F1, NT1, Kt1, Wp1);
    k_pack_w<<<nblk((long)Kt2 * NT2 * 512, BS), BS, 0, stream>>>(W2, F1, F2, NT2, Kt2, Wp2);

    const int rowTiles32 = (N + 31) / 32;  // 3125

    // ---- layer 1: h1 = x @ W1 (WMMA), aggregate, bias+relu ----
    k_wmma_gemm<16, 8><<<rowTiles32, 8 * 32, 0, stream>>>(x, N, F_IN, Wp1, h1, F1);
    k_edge_agg<<<nblk((long)E * (F1 / 4), BS), BS, 0, stream>>>(src, dst, dis, h1, agg1, E, F1);
    k_bias_relu<<<nblk((long)N * F1, BS), BS, 0, stream>>>(agg1, h1, dis, b1, N, F1);

    // ---- layer 2: h2 = h1 @ W2 (WMMA), aggregate, bias+relu ----
    k_wmma_gemm<4, 2><<<rowTiles32, 2 * 32, 0, stream>>>(h1, N, F1, Wp2, h2, F2);
    k_edge_agg<<<nblk((long)E * (F2 / 4), BS), BS, 0, stream>>>(src, dst, dis, h2, agg2, E, F2);
    k_bias_relu<<<nblk((long)N * F2, BS), BS, 0, stream>>>(agg2, h2, dis, b2, N, F2);

    // ---- layer 3: tiny dense + aggregate + log_softmax ----
    k_dense3<<<nblk(N, BS), BS, 0, stream>>>(h2, W3, h3, N);
    k_edge_agg3<<<nblk(E, BS), BS, 0, stream>>>(src, dst, dis, h3, agg3, E);
    k_finalize<<<nblk(N, BS), BS, 0, stream>>>(agg3, h3, dis, b3, out, N);
}
